// BACAM_40870908788791
// MI455X (gfx1250) — compile-verified, matrix operand order checked by reading, not run on hardware
//
#include <hip/hip_runtime.h>
#include <hip/hip_bf16.h>
#include <math.h>

// ---------------- problem constants ----------------
#define BB   8
#define HH_  64
#define WW_  64
#define CC   256
#define PP   32768      // BB * HH_ * WW_   (pixel rows of the big GEMMs)
#define GG   8
#define CGD  32         // CC / GG
#define DQK  176
#define PGG  262144     // 64 groups * 4096 pixels (ba-branch "pixels")

typedef __bf16 bf16;
typedef __attribute__((ext_vector_type(16))) __bf16 v16bf;
typedef __attribute__((ext_vector_type(8)))  float  v8f;
typedef __attribute__((ext_vector_type(4)))  unsigned int v4u;

union Frag { v16bf v; v4u q[2]; };

__device__ __forceinline__ bf16 f2bf(float f) {
  unsigned u = __float_as_uint(f);
  unsigned r = u + 0x7fffu + ((u >> 16) & 1u);     // round-to-nearest-even
  unsigned short h = (unsigned short)(r >> 16);
  return __builtin_bit_cast(bf16, h);
}
__device__ __forceinline__ float geluf(float x) {
  return 0.5f * x * (1.0f + erff(x * 0.70710678118654752f));
}
__device__ __forceinline__ float sigm(float x) {
  return 1.0f / (1.0f + expf(-x));
}

// Load one 16x16x32 bf16 WMMA fragment half-pair: two contiguous 16B runs.
// p must already point at  base + row*ld + kbase + sel8.
__device__ __forceinline__ v16bf load_frag(const bf16* p) {
  Frag f;
  f.q[0] = *reinterpret_cast<const v4u*>(p);
  f.q[1] = *reinterpret_cast<const v4u*>(p + 16);
  return f.v;
}

// gfx1250 async copy: 16B global -> LDS, tracked by ASYNCcnt.
// VDST operand is the LDS byte offset = low 32 bits of the flat shared address.
__device__ __forceinline__ void async_g2l_b128(const void* gptr, void* lptr) {
  unsigned loff = (unsigned)(uintptr_t)lptr;
  asm volatile("global_load_async_to_lds_b128 %0, %1, off"
               :: "v"(loff), "v"(gptr) : "memory");
}
__device__ __forceinline__ void wait_asynccnt0() {
  asm volatile("s_wait_asynccnt 0" ::: "memory");
}

// ---------------- weight staging ----------------
__global__ void k_cvt_bf16(const float* __restrict__ src, bf16* __restrict__ dst, int n) {
  int i = blockIdx.x * 256 + threadIdx.x;
  if (i < n) dst[i] = f2bf(src[i]);
}
// attn2_w (200x176) -> zero-padded bf16 [208][192]
__global__ void k_stage_wa2(const float* __restrict__ w, bf16* __restrict__ dst) {
  int i = blockIdx.x * 256 + threadIdx.x;
  if (i >= 208 * 192) return;
  int d = i / 192, k = i % 192;
  float v = (d < 200 && k < 176) ? w[d * 176 + k] : 0.0f;
  dst[i] = f2bf(v);
}
__global__ void k_zero_f32(float* __restrict__ p, int n) {
  int i = blockIdx.x * 256 + threadIdx.x;
  if (i < n) p[i] = 0.0f;
}

// ---------------- K1: depthwise 3x3 + relu + layernorm (NHWC) ----------------
template <int STORE_BF>
__global__ void k_proj(const float* __restrict__ x, const float* __restrict__ cw,
                       const float* __restrict__ cb, const float* __restrict__ lw,
                       const float* __restrict__ lb, bf16* __restrict__ obf,
                       float* __restrict__ of32) {
  __shared__ float red[256];
  int p = blockIdx.x;                 // pixel 0..32767
  int b = p >> 12, hw = p & 4095, h = hw >> 6, w = hw & 63;
  int c = threadIdx.x;
  float acc = cb[c];
#pragma unroll
  for (int dh = 0; dh < 3; dh++) {
    int r = h + dh - 1; if (r < 0 || r > 63) continue;
#pragma unroll
    for (int dw = 0; dw < 3; dw++) {
      int cl = w + dw - 1; if (cl < 0 || cl > 63) continue;
      acc += x[(size_t)(((b << 12) + (r << 6) + cl)) * CC + c] * cw[c * 9 + dh * 3 + dw];
    }
  }
  float t = fmaxf(acc, 0.0f);
  // block LN over the 256 channels (sum)
  red[c] = t; __syncthreads();
  for (int off = 128; off > 0; off >>= 1) { if (c < off) red[c] += red[c + off]; __syncthreads(); }
  float mean = red[0] * (1.0f / 256.0f); __syncthreads();
  float d = t - mean;
  red[c] = d * d; __syncthreads();
  for (int off = 128; off > 0; off >>= 1) { if (c < off) red[c] += red[c + off]; __syncthreads(); }
  float var = red[0] * (1.0f / 256.0f);
  float y = d * rsqrtf(var + 1e-5f) * lw[c] + lb[c];
  if (STORE_BF) obf[(size_t)p * CC + c] = f2bf(y);
  else          of32[(size_t)p * CC + c] = y;
}

// ---------------- K2: fused conv_qk GEMM on q,k + hadamard*scale (WMMA) ----------------
__global__ void k_had(const bf16* __restrict__ qln, const bf16* __restrict__ kln,
                      const bf16* __restrict__ wqk, const float* __restrict__ bias,
                      float* __restrict__ had) {
  const int NT = 11;                                   // 176/16
  int wave = (blockIdx.x * blockDim.x + threadIdx.x) >> 5;
  int lane = threadIdx.x & 31;
  int mt = wave / NT, nt = wave % NT;
  int m = mt * 16 + (lane & 15);
  int n = nt * 16 + (lane & 15);
  int sel = (lane & 16) ? 8 : 0;
  v8f accq = {0,0,0,0,0,0,0,0};
  v8f acck = {0,0,0,0,0,0,0,0};
#pragma unroll
  for (int kb = 0; kb < 256; kb += 32) {
    v16bf bfrag = load_frag(wqk + (size_t)n * 256 + kb + sel);
    v16bf aq    = load_frag(qln + (size_t)m * 256 + kb + sel);
    v16bf ak    = load_frag(kln + (size_t)m * 256 + kb + sel);
    accq = __builtin_amdgcn_wmma_f32_16x16x32_bf16(false, aq, false, bfrag, (short)0, accq, false, false);
    acck = __builtin_amdgcn_wmma_f32_16x16x32_bf16(false, ak, false, bfrag, (short)0, acck, false, false);
  }
  float bn = bias[n];
  const float SCALE = 0.17677669529663687f;            // 32^-0.5
#pragma unroll
  for (int i = 0; i < 8; i++) {
    int row = mt * 16 + i + ((lane & 16) ? 8 : 0);
    float qv = accq[i] + bn, kv = acck[i] + bn;
    had[(size_t)row * DQK + n] = qv * kv * SCALE;
  }
}

// ---------------- K3: grouped 5x5 conv + gelu -> bf16 (K-padded to 192) ----------------
__global__ void k_attn1(const float* __restrict__ had, const float* __restrict__ w1,
                        const float* __restrict__ b1, bf16* __restrict__ a1g) {
  int id = blockIdx.x * 256 + threadIdx.x;             // 32768*192
  int oc = id % 192, p = id / 192;
  if (oc >= DQK) { a1g[(size_t)p * 192 + oc] = f2bf(0.0f); return; }
  int b = p >> 12, hw = p & 4095, h = hw >> 6, w = hw & 63;
  int icb = (oc >> 3) << 3;                            // 8-channel group base
  float acc = b1[oc];
  for (int dh = 0; dh < 5; dh++) {
    int r = h + dh - 2; if (r < 0 || r > 63) continue;
    for (int dw = 0; dw < 5; dw++) {
      int cl = w + dw - 2; if (cl < 0 || cl > 63) continue;
      const float* hp = had + (size_t)((b << 12) + (r << 6) + cl) * DQK + icb;
      const float* wp = w1 + (size_t)oc * 200 + dh * 5 + dw;
#pragma unroll
      for (int ic = 0; ic < 8; ic++) acc += hp[ic] * wp[ic * 25];
    }
  }
  a1g[(size_t)p * 192 + oc] = f2bf(geluf(acc));
}

// ---------------- K4: attn2 GEMM 192x208 (WMMA) ----------------
__global__ void k_attn2(const bf16* __restrict__ a1g, const bf16* __restrict__ wa2,
                        const float* __restrict__ bias, float* __restrict__ araw) {
  const int NT = 13;                                   // 208/16
  int wave = (blockIdx.x * blockDim.x + threadIdx.x) >> 5;
  int lane = threadIdx.x & 31;
  int mt = wave / NT, nt = wave % NT;
  int m = mt * 16 + (lane & 15);
  int n = nt * 16 + (lane & 15);
  int sel = (lane & 16) ? 8 : 0;
  v8f acc = {0,0,0,0,0,0,0,0};
#pragma unroll
  for (int kb = 0; kb < 192; kb += 32) {
    v16bf bfrag = load_frag(wa2 + (size_t)n * 192 + kb + sel);
    v16bf afrag = load_frag(a1g + (size_t)m * 192 + kb + sel);
    acc = __builtin_amdgcn_wmma_f32_16x16x32_bf16(false, afrag, false, bfrag, (short)0, acc, false, false);
  }
  if (n < 200) {
    float bn = bias[n];
#pragma unroll
    for (int i = 0; i < 8; i++) {
      int row = mt * 16 + i + ((lane & 16) ? 8 : 0);
      araw[(size_t)row * 208 + n] = acc[i] + bn;
    }
  }
}

// ---------------- K5: per-head softmax over 25 taps ----------------
__global__ void k_softmax(const float* __restrict__ araw, float* __restrict__ hat) {
  int id = blockIdx.x * 256 + threadIdx.x;             // 32768*8
  int g = id & 7, p = id >> 3;
  const float* a = araw + (size_t)p * 208 + g * 25;
  float mx = -1e30f;
#pragma unroll
  for (int i = 0; i < 25; i++) mx = fmaxf(mx, a[i]);
  float e[25], s = 0.0f;
#pragma unroll
  for (int i = 0; i < 25; i++) { e[i] = expf(a[i] - mx); s += e[i]; }
  float inv = 1.0f / s;
  float* o = hat + (size_t)p * 200 + g * 25;
#pragma unroll
  for (int i = 0; i < 25; i++) o[i] = e[i] * inv;
}

// ---------------- K6a: depthwise 7x7 on vg ----------------
__global__ void k_dw7(const float* __restrict__ vln, const float* __restrict__ w7,
                      const float* __restrict__ b7, float* __restrict__ xd) {
  int id = blockIdx.x * 256 + threadIdx.x;             // 64*4096*32
  int cg = id & 31, pg = id >> 5;
  int bg = pg >> 12, hw = pg & 4095, h = hw >> 6, w = hw & 63;
  int b = bg >> 3, g = bg & 7, ch = g * 32 + cg;
  float acc = b7[cg];
  for (int i = 0; i < 7; i++) {
    int r = h + i - 3; if (r < 0 || r > 63) continue;
    for (int j = 0; j < 7; j++) {
      int cl = w + j - 3; if (cl < 0 || cl > 63) continue;
      acc += vln[(size_t)((b << 12) + (r << 6) + cl) * CC + ch] * w7[cg * 49 + i * 7 + j];
    }
  }
  xd[(size_t)pg * 32 + cg] = acc;
}

// ---------------- K6b: vertical 5x1 & 3x1 convs + gelu ----------------
__global__ void k_vconv(const float* __restrict__ xd, const float* __restrict__ w5,
                        const float* __restrict__ b5, const float* __restrict__ w3,
                        const float* __restrict__ b3, float* __restrict__ g51,
                        float* __restrict__ g31) {
  int id = blockIdx.x * 256 + threadIdx.x;
  int cg = id & 31, pg = id >> 5;
  int bg = pg >> 12, hw = pg & 4095, h = hw >> 6, w = hw & 63;
  float a5 = b5[cg];
#pragma unroll
  for (int t = 0; t < 5; t++) {
    int r = h + t - 2; if (r < 0 || r > 63) continue;
    a5 += xd[(size_t)((bg << 12) + (r << 6) + w) * 32 + cg] * w5[cg * 5 + t];
  }
  float a3 = b3[cg];
#pragma unroll
  for (int t = 0; t < 3; t++) {
    int r = h + t - 1; if (r < 0 || r > 63) continue;
    a3 += xd[(size_t)((bg << 12) + (r << 6) + w) * 32 + cg] * w3[cg * 3 + t];
  }
  g51[(size_t)pg * 32 + cg] = geluf(a5);
  g31[(size_t)pg * 32 + cg] = geluf(a3);
}

// ---------------- K6c: horizontal convs + sigmoid + gate + pw(32x32) + BN stats ----
// pw_w (4KB) is staged into LDS with gfx1250 async-to-LDS DMA (ASYNCcnt).
__global__ void k_pw1(const float* __restrict__ g51, const float* __restrict__ g31,
                      const float* __restrict__ vln, const float* __restrict__ w15,
                      const float* __restrict__ b15, const float* __restrict__ w13,
                      const float* __restrict__ b13, const float* __restrict__ pww,
                      const float* __restrict__ pwb, float* __restrict__ t1,
                      float* __restrict__ stats) {
  __shared__ float wlds[1024];                 // 32x32 pointwise weights
  __shared__ float xf[8][32];
  __shared__ float tv[8][32];
  async_g2l_b128(pww + threadIdx.x * 4, &wlds[threadIdx.x * 4]);

  int li = threadIdx.x >> 5, cg = threadIdx.x & 31;
  int pg = blockIdx.x * 8 + li;
  int bg = pg >> 12, hw = pg & 4095, h = hw >> 6, w = hw & 63;
  int b = bg >> 3, g = bg & 7, ch = g * 32 + cg;
  float a5 = b15[cg];
#pragma unroll
  for (int t = 0; t < 5; t++) {
    int cl = w + t - 2; if (cl < 0 || cl > 63) continue;
    a5 += g51[(size_t)((bg << 12) + (h << 6) + cl) * 32 + cg] * w15[cg * 5 + t];
  }
  float a3 = b13[cg];
#pragma unroll
  for (int t = 0; t < 3; t++) {
    int cl = w + t - 1; if (cl < 0 || cl > 63) continue;
    a3 += g31[(size_t)((bg << 12) + (h << 6) + cl) * 32 + cg] * w13[cg * 3 + t];
  }
  float vv = vln[(size_t)((b << 12) + hw) * CC + ch];
  xf[li][cg] = vv * (sigm(a5) + sigm(a3)) + vv;
  wait_asynccnt0();
  __syncthreads();
  float acc = pwb[cg];
#pragma unroll
  for (int ic = 0; ic < 32; ic++) acc += wlds[cg * 32 + ic] * xf[li][ic];
  t1[(size_t)pg * 32 + cg] = acc;
  tv[li][cg] = acc;
  __syncthreads();
  if (li == 0) {
    float s = 0.0f, s2 = 0.0f;
#pragma unroll
    for (int j = 0; j < 8; j++) { float v = tv[j][cg]; s += v; s2 += v * v; }
    atomicAdd(&stats[cg], s);
    atomicAdd(&stats[32 + cg], s2);
  }
}

// ---------------- K6d/K6f: finalize BN -> affine (a, b') ----------------
__global__ void k_bn_final(const float* __restrict__ stats, const float* __restrict__ bnw,
                           const float* __restrict__ bnb, float* __restrict__ ab) {
  int c = threadIdx.x;
  if (c < 32) {
    const float inv = 1.0f / 262144.0f;
    float m = stats[c] * inv;
    float v = stats[32 + c] * inv - m * m;
    float a = rsqrtf(v + 1e-5f) * bnw[c];
    ab[c] = a;
    ab[32 + c] = bnb[c] - m * a;
  }
}

// ---------------- K6e: bn+gelu(t1) -> pw -> t2 + stats2 ----------------
__global__ void k_pw2(const float* __restrict__ t1, const float* __restrict__ ab1,
                      const float* __restrict__ pww, const float* __restrict__ pwb,
                      float* __restrict__ t2, float* __restrict__ stats2) {
  __shared__ float wlds[1024];
  __shared__ float s1[8][32];
  __shared__ float tv[8][32];
  async_g2l_b128(pww + threadIdx.x * 4, &wlds[threadIdx.x * 4]);

  int li = threadIdx.x >> 5, cg = threadIdx.x & 31;
  int pg = blockIdx.x * 8 + li;
  float tval = t1[(size_t)pg * 32 + cg];
  s1[li][cg] = geluf(tval * ab1[cg] + ab1[32 + cg]);
  wait_asynccnt0();
  __syncthreads();
  float acc = pwb[cg];
#pragma unroll
  for (int ic = 0; ic < 32; ic++) acc += wlds[cg * 32 + ic] * s1[li][ic];
  t2[(size_t)pg * 32 + cg] = acc;
  tv[li][cg] = acc;
  __syncthreads();
  if (li == 0) {
    float s = 0.0f, s2 = 0.0f;
#pragma unroll
    for (int j = 0; j < 8; j++) { float v = tv[j][cg]; s += v; s2 += v * v; }
    atomicAdd(&stats2[cg], s);
    atomicAdd(&stats2[32 + cg], s2);
  }
}

// ---------------- K6g: feat = gelu(bn2(t2)) ----------------
__global__ void k_feat(const float* __restrict__ t2, const float* __restrict__ ab2,
                       float* __restrict__ feat) {
  int id = blockIdx.x * 256 + threadIdx.x;
  int cg = id & 31;
  float v = t2[id];
  feat[id] = geluf(v * ab2[cg] + ab2[32 + cg]);
}

// ---------------- K7: ghost aggregation -> NHWC bf16 ----------------
__global__ void k_ghost(const float* __restrict__ feat, const float* __restrict__ hat,
                        const float* __restrict__ ghost, bf16* __restrict__ outc) {
  int id = blockIdx.x * 256 + threadIdx.x;             // 64*4096*32
  int cg = id & 31, pg = id >> 5;
  int bg = pg >> 12, hw = pg & 4095, h = hw >> 6, w = hw & 63;
  int b = bg >> 3, g = bg & 7, ch = g * 32 + cg;
  const float* gm = ghost + (size_t)ch * 25;
  const float* ga = ghost + 256 * 25 + (size_t)ch * 25;
  const float* ha = hat + (size_t)((b << 12) + hw) * 200 + g * 25;
  float acc = 0.0f;
#pragma unroll
  for (int idx = 0; idx < 25; idx++) {
    int i = idx / 5, j = idx % 5;
    int r = h + i - 2, cl = w + j - 2;
    float patch = (r >= 0 && r < 64 && cl >= 0 && cl < 64)
                      ? feat[(size_t)((bg << 12) + (r << 6) + cl) * 32 + cg] : 0.0f;
    acc += patch * (gm[idx] * ha[idx] + ga[idx]);
  }
  outc[(size_t)((b << 12) + hw) * CC + ch] = f2bf(acc);
}

// ---------------- K8: post einsum GEMM 256x256 + bias (WMMA) ----------------
__global__ void k_post(const bf16* __restrict__ outc, const bf16* __restrict__ wpost,
                       const float* __restrict__ bias, float* __restrict__ y) {
  const int NT = 16;
  int wave = (blockIdx.x * blockDim.x + threadIdx.x) >> 5;
  int lane = threadIdx.x & 31;
  int mt = wave / NT, nt = wave % NT;
  int m = mt * 16 + (lane & 15);
  int n = nt * 16 + (lane & 15);
  int sel = (lane & 16) ? 8 : 0;
  v8f acc = {0,0,0,0,0,0,0,0};
#pragma unroll
  for (int kb = 0; kb < 256; kb += 32) {
    if (kb + 64 < 256) __builtin_prefetch(outc + (size_t)m * 256 + kb + 64, 0, 0);
    v16bf bfrag = load_frag(wpost + (size_t)n * 256 + kb + sel);
    v16bf afrag = load_frag(outc + (size_t)m * 256 + kb + sel);
    acc = __builtin_amdgcn_wmma_f32_16x16x32_bf16(false, afrag, false, bfrag, (short)0, acc, false, false);
  }
  float bn = bias[n];
#pragma unroll
  for (int i = 0; i < 8; i++) {
    int row = mt * 16 + i + ((lane & 16) ? 8 : 0);
    y[(size_t)row * CC + n] = acc[i] + bn;
  }
}

// ---------------- host orchestration ----------------
extern "C" void kernel_launch(void* const* d_in, const int* in_sizes, int n_in,
                              void* d_out, int out_size, void* d_ws, size_t ws_size,
                              hipStream_t stream) {
  (void)in_sizes; (void)n_in; (void)out_size; (void)ws_size;
  const float* x        = (const float*)d_in[0];
  const float* conv_q_w = (const float*)d_in[3];
  const float* conv_q_b = (const float*)d_in[4];
  const float* conv_k_w = (const float*)d_in[5];
  const float* conv_k_b = (const float*)d_in[6];
  const float* conv_v_w = (const float*)d_in[7];
  const float* conv_v_b = (const float*)d_in[8];
  const float* ln_q_w   = (const float*)d_in[9];
  const float* ln_q_b   = (const float*)d_in[10];
  const float* ln_k_w   = (const float*)d_in[11];
  const float* ln_k_b   = (const float*)d_in[12];
  const float* ln_v_w   = (const float*)d_in[13];
  const float* ln_v_b   = (const float*)d_in[14];
  const float* conv_qk_w = (const float*)d_in[15];
  const float* conv_qk_b = (const float*)d_in[16];
  const float* attn1_w  = (const float*)d_in[17];
  const float* attn1_b  = (const float*)d_in[18];
  const float* attn2_w  = (const float*)d_in[19];
  const float* attn2_b  = (const float*)d_in[20];
  const float* ghost    = (const float*)d_in[21];
  const float* post_w   = (const float*)d_in[22];
  const float* post_b   = (const float*)d_in[23];
  const float* dw7w = (const float*)d_in[24];
  const float* dw7b = (const float*)d_in[25];
  const float* s51w = (const float*)d_in[26];
  const float* s51b = (const float*)d_in[27];
  const float* s15w = (const float*)d_in[28];
  const float* s15b = (const float*)d_in[29];
  const float* s31w = (const float*)d_in[30];
  const float* s31b = (const float*)d_in[31];
  const float* s13w = (const float*)d_in[32];
  const float* s13b = (const float*)d_in[33];
  const float* pww  = (const float*)d_in[34];
  const float* pwb  = (const float*)d_in[35];
  const float* bnw  = (const float*)d_in[36];
  const float* bnb  = (const float*)d_in[37];

  char* ws = (char*)d_ws;
  size_t cur = 0;
  auto alloc = [&](size_t n) { size_t r = cur; cur += n; cur = (cur + 255) & ~(size_t)255; return r; };

  size_t o_vln   = alloc((size_t)PP * CC * 4);     // f32, lives K1 -> k_pw1
  size_t o_slabA = alloc((size_t)PP * CC * 4);     // qln+kln(bf16) -> xd(f32) -> t2(f32)
  size_t o_slabB = alloc((size_t)PP * CC * 4);     // had(f32) -> g51(f32) -> feat(f32)
  size_t o_slabC = alloc(39845888);                // a1g(bf16)+attn_raw(f32) -> g31(f32) -> outc(bf16)
  size_t o_hat   = alloc((size_t)PP * 200 * 4);    // softmaxed attention
  size_t o_t1    = alloc((size_t)PGG * 32 * 4);
  size_t o_wqk   = alloc((size_t)DQK * 256 * 2);
  size_t o_wa2   = alloc((size_t)208 * 192 * 2);
  size_t o_wpost = alloc((size_t)256 * 256 * 2);
  size_t o_stats = alloc(256 * 4);                 // stats1[64], stats2[64], ab1[64], ab2[64]

  float* vln   = (float*)(ws + o_vln);
  bf16*  qln   = (bf16*)(ws + o_slabA);
  bf16*  kln   = (bf16*)(ws + o_slabA + (size_t)PP * CC * 2);
  float* xd    = (float*)(ws + o_slabA);
  float* t2    = (float*)(ws + o_slabA);
  float* had   = (float*)(ws + o_slabB);
  float* g51   = (float*)(ws + o_slabB);
  float* feat  = (float*)(ws + o_slabB);
  bf16*  a1g   = (bf16*)(ws + o_slabC);
  float* araw  = (float*)(ws + o_slabC + (size_t)PP * 192 * 2);
  float* g31   = (float*)(ws + o_slabC);
  bf16*  outc  = (bf16*)(ws + o_slabC);
  float* hat   = (float*)(ws + o_hat);
  float* t1    = (float*)(ws + o_t1);
  bf16*  wqk   = (bf16*)(ws + o_wqk);
  bf16*  wa2   = (bf16*)(ws + o_wa2);
  bf16*  wpost = (bf16*)(ws + o_wpost);
  float* stats1 = (float*)(ws + o_stats);
  float* stats2 = stats1 + 64;
  float* ab1    = stats1 + 128;
  float* ab2    = stats1 + 192;
  float* yout   = (float*)d_out;

  // stage weights (bf16) + zero BN stats
  k_cvt_bf16<<<(DQK * 256 + 255) / 256, 256, 0, stream>>>(conv_qk_w, wqk, DQK * 256);
  k_cvt_bf16<<<(256 * 256 + 255) / 256, 256, 0, stream>>>(post_w, wpost, 256 * 256);
  k_stage_wa2<<<(208 * 192 + 255) / 256, 256, 0, stream>>>(attn2_w, wa2);
  k_zero_f32<<<1, 256, 0, stream>>>(stats1, 128);

  // K1: projections
  k_proj<1><<<PP, 256, 0, stream>>>(x, conv_q_w, conv_q_b, ln_q_w, ln_q_b, qln, nullptr);
  k_proj<1><<<PP, 256, 0, stream>>>(x, conv_k_w, conv_k_b, ln_k_w, ln_k_b, kln, nullptr);
  k_proj<0><<<PP, 256, 0, stream>>>(x, conv_v_w, conv_v_b, ln_v_w, ln_v_b, nullptr, vln);

  // K2: fused qk GEMM + hadamard  (2048*11 tiles, 8 waves/block)
  k_had<<<2816, 256, 0, stream>>>(qln, kln, wqk, conv_qk_b, had);
  // K3: grouped 5x5 + gelu -> bf16
  k_attn1<<<(PP * 192) / 256, 256, 0, stream>>>(had, attn1_w, attn1_b, a1g);
  // K4: attn2 GEMM (2048*13 tiles)
  k_attn2<<<3328, 256, 0, stream>>>(a1g, wa2, attn2_b, araw);
  // K5: softmax
  k_softmax<<<(PP * 8) / 256, 256, 0, stream>>>(araw, hat);

  // K6: ba branch
  k_dw7<<<(PGG * 32) / 256, 256, 0, stream>>>(vln, dw7w, dw7b, xd);
  k_vconv<<<(PGG * 32) / 256, 256, 0, stream>>>(xd, s51w, s51b, s31w, s31b, g51, g31);
  k_pw1<<<PGG / 8, 256, 0, stream>>>(g51, g31, vln, s15w, s15b, s13w, s13b, pww, pwb, t1, stats1);
  k_bn_final<<<1, 32, 0, stream>>>(stats1, bnw, bnb, ab1);
  k_pw2<<<PGG / 8, 256, 0, stream>>>(t1, ab1, pww, pwb, t2, stats2);
  k_bn_final<<<1, 32, 0, stream>>>(stats2, bnw, bnb, ab2);
  k_feat<<<(PGG * 32) / 256, 256, 0, stream>>>(t2, ab2, feat);

  // K7: ghost aggregation
  k_ghost<<<(PGG * 32) / 256, 256, 0, stream>>>(feat, hat, ghost, outc);
  // K8: post einsum GEMM (2048*16 tiles)
  k_post<<<4096, 256, 0, stream>>>(outc, wpost, post_b, yout);
}
